// NCABlock_41661182771217
// MI455X (gfx1250) — matrix operand order repaired and checked by reading, not compiled, not run on hardware
//
#include <hip/hip_runtime.h>
#include <cmath>

// Problem constants (from reference): N=4, C=128, H=W=64, HEADS=4, KS=7, D=32
#define MTOT 16384   // N*H*W token rows
#define CCH  128
#define HWW  4096
#define NB   4
#define NHEADS 4
#define DH   32
#define KSZ  7
#define KK2  49

typedef __attribute__((ext_vector_type(16))) __bf16 v16bf;
typedef __attribute__((ext_vector_type(8)))  __bf16 v8bf;
typedef __attribute__((ext_vector_type(8)))  float  v8f;

__device__ __forceinline__ unsigned short f2bf(float f) {
    union { float f; unsigned u; } v; v.f = f;
    unsigned r = v.u + 0x7fffu + ((v.u >> 16) & 1u);   // round-to-nearest-even
    return (unsigned short)(r >> 16);
}

__device__ __forceinline__ float gelu_exact(float x) {
    return 0.5f * x * (1.0f + erff(x * 0.70710678118654752f));
}

// ---- weight convert f32 (K,N) -> bf16 transposed (N,K) -----------------
__global__ void k_convert_wT(const float* __restrict__ in,
                             unsigned short* __restrict__ out, int K, int N) {
    int gid = blockIdx.x * blockDim.x + threadIdx.x;
    if (gid >= K * N) return;
    int k = gid / N, n = gid - k * N;
    out[(size_t)n * K + k] = f2bf(in[gid]);
}

// ---- LN1 fused with NCHW -> (M,C) transpose ----------------------------
__global__ void k_ln1(const float* __restrict__ x, const float* __restrict__ w,
                      const float* __restrict__ b, float* __restrict__ xT,
                      unsigned short* __restrict__ xnh) {
    int p = blockIdx.x, c = threadIdx.x;
    int n = p >> 12, hw = p & 4095;
    float v = x[(((size_t)n * CCH + c) << 12) + hw];
    xT[(size_t)p * CCH + c] = v;
    __shared__ float s1[CCH], s2[CCH];
    s1[c] = v; s2[c] = v * v; __syncthreads();
    for (int o = 64; o > 0; o >>= 1) {
        if (c < o) { s1[c] += s1[c + o]; s2[c] += s2[c + o]; }
        __syncthreads();
    }
    float mu  = s1[0] * (1.0f / CCH);
    float var = s2[0] * (1.0f / CCH) - mu * mu;
    float y = (v - mu) * rsqrtf(var + 1e-5f) * w[c] + b[c];
    xnh[(size_t)p * CCH + c] = f2bf(y);
}

// ---- generic bf16 WMMA GEMM:  out = epi(A[M,K] @ B[K,N] + bias) --------
// A: bf16 row-major (M,K).  BT: bf16 (N,K) (pre-transposed weights).
// flags: bit0 = GELU, bit1 = add f32 residual res[M,N], bit2 = store outF
//        transposed as NCHW (N=CCH assumed).
// Block = 256 threads = 8 waves arranged 4x2; each wave owns a 32x32 output
// tile (2x2 WMMA sub-tiles, 4 f32 accumulators) -> block tile 128x64.
// Per K-step: 2 A frags + 2 B frags (8x b128) feed 4 WMMAs = 2 loads/WMMA.
// Fragment layout per CDNA5 ISA §7.12.2:
//   A 16x32 bf16: lane L (row = L&15): lanes 0-15 hold K {k0..k0+7, k0+16..k0+23},
//                 lanes 16-31 hold K {k0+8..k0+15, k0+24..k0+31}.
//   B 32x16 bf16: lane L (col = L&15): same K split as A.
//   C/D f32 16x16: lane L col = L&15, VGPR r -> row r + 8*(L>>4).
__global__ __launch_bounds__(256) void k_wmma_gemm(
        const unsigned short* __restrict__ A, const unsigned short* __restrict__ BT,
        const float* __restrict__ bias, const float* __restrict__ res,
        float* __restrict__ outF, unsigned short* __restrict__ outH,
        int K, int N, int flags) {
    int lane = threadIdx.x & 31, wave = threadIdx.x >> 5;
    int l16 = lane & 15, lh = lane >> 4;
    int mBase = blockIdx.y * 128 + (wave >> 1) * 32;
    int nBase = blockIdx.x * 64  + (wave & 1) * 32;
    // column load bases clamped into range (stores are guarded separately)
    int nb0 = nBase      <= N - 16 ? nBase      : N - 16;
    int nb1 = nBase + 16 <= N - 16 ? nBase + 16 : N - 16;

    const unsigned short* a0 = A  + (size_t)(mBase + l16)      * K + lh * 8;
    const unsigned short* a1 = A  + (size_t)(mBase + 16 + l16) * K + lh * 8;
    const unsigned short* b0 = BT + (size_t)(nb0 + l16)        * K + lh * 8;
    const unsigned short* b1 = BT + (size_t)(nb1 + l16)        * K + lh * 8;

    auto ldA = [&](const unsigned short* p) -> v16bf {
        v8bf lo = *(const v8bf*)p;
        v8bf hi = *(const v8bf*)(p + 16);
        v16bf f;
#pragma unroll
        for (int i = 0; i < 8; i++) { f[i] = lo[i]; f[i + 8] = hi[i]; }
        return f;
    };

    v8f acc00 = {}, acc01 = {}, acc10 = {}, acc11 = {};
    for (int k0 = 0; k0 < K; k0 += 32) {
        v16bf af0 = ldA(a0 + k0);
        v16bf af1 = ldA(a1 + k0);
        v16bf bf0 = ldA(b0 + k0);
        v16bf bf1 = ldA(b1 + k0);
        acc00 = __builtin_amdgcn_wmma_f32_16x16x32_bf16(false, af0, false, bf0,
                                                        (short)0, acc00, false, false);
        acc01 = __builtin_amdgcn_wmma_f32_16x16x32_bf16(false, af0, false, bf1,
                                                        (short)0, acc01, false, false);
        acc10 = __builtin_amdgcn_wmma_f32_16x16x32_bf16(false, af1, false, bf0,
                                                        (short)0, acc10, false, false);
        acc11 = __builtin_amdgcn_wmma_f32_16x16x32_bf16(false, af1, false, bf1,
                                                        (short)0, acc11, false, false);
    }

    auto epi = [&](v8f acc, int mB, int nB) {
        if (nB >= N) return;                 // clamped duplicate tile: no store
        int col = nB + l16;
        float bc = bias ? bias[col] : 0.0f;
#pragma unroll
        for (int r = 0; r < 8; r++) {
            int rr = mB + r + lh * 8;
            float v = acc[r] + bc;
            if (flags & 1) v = gelu_exact(v);
            if (flags & 2) v += res[(size_t)rr * N + col];
            if (outH) outH[(size_t)rr * N + col] = f2bf(v);
            if (outF) {
                if (flags & 4) {
                    int nn = rr >> 12, hw = rr & 4095;
                    outF[(((size_t)nn * CCH + col) << 12) + hw] = v;
                } else {
                    outF[(size_t)rr * N + col] = v;
                }
            }
        }
    };
    epi(acc00, mBase,      nBase);
    epi(acc01, mBase,      nBase + 16);
    epi(acc10, mBase + 16, nBase);
    epi(acc11, mBase + 16, nBase + 16);
}

// ---- 7x7 neighborhood attention (one thread per pixel-head) ------------
__global__ __launch_bounds__(256) void k_na2d(const float* __restrict__ qkv,
                                              const float* __restrict__ rpb,
                                              unsigned short* __restrict__ attnh) {
    __shared__ float slog[256 * KK2];
    int tid = threadIdx.x;
    int gid = blockIdx.x * 256 + tid;
    int head = gid & 3;
    int pix  = gid >> 2;
    int n = pix >> 12, rem = pix & 4095, i = rem >> 6, j = rem & 63;

    const float4* qp = (const float4*)(qkv + (size_t)pix * 384 + head * DH);
    float q[DH];
#pragma unroll
    for (int d4 = 0; d4 < 8; d4++) {
        float4 t = qp[d4];
        q[4*d4+0] = t.x * 0.17677669529663687f;   // D^-0.5
        q[4*d4+1] = t.y * 0.17677669529663687f;
        q[4*d4+2] = t.z * 0.17677669529663687f;
        q[4*d4+3] = t.w * 0.17677669529663687f;
    }
    int ni = i - 3; if (ni < 0) ni = 0; if (ni > 57) ni = 57;   // H-KS = 57
    int nj = j - 3; if (nj < 0) nj = 0; if (nj > 57) nj = 57;

    float* lg = slog + tid * KK2;
    float mx = -3.0e38f;
    int ai = 0, aj = 0;
    for (int a = 0; a < KK2; a++) {
        int ki = ni + ai, kj = nj + aj;
        size_t kpix = ((size_t)(n * 64 + ki) << 6) + kj;
        const float4* kp = (const float4*)(qkv + kpix * 384 + 128 + head * DH);
        float s = 0.0f;
#pragma unroll
        for (int d4 = 0; d4 < 8; d4++) {
            float4 kv = kp[d4];
            s += q[4*d4]*kv.x + q[4*d4+1]*kv.y + q[4*d4+2]*kv.z + q[4*d4+3]*kv.w;
        }
        s += rpb[head * 169 + (ki - i + 6) * 13 + (kj - j + 6)];
        lg[a] = s; if (s > mx) mx = s;
        aj++; if (aj == KSZ) { aj = 0; ai++; }
    }
    float sum = 0.0f;
    for (int a = 0; a < KK2; a++) { float e = expf(lg[a] - mx); lg[a] = e; sum += e; }
    float inv = 1.0f / sum;

    float o[DH];
#pragma unroll
    for (int d = 0; d < DH; d++) o[d] = 0.0f;
    ai = 0; aj = 0;
    for (int a = 0; a < KK2; a++) {
        int ki = ni + ai, kj = nj + aj;
        size_t vpix = ((size_t)(n * 64 + ki) << 6) + kj;
        const float4* vp = (const float4*)(qkv + vpix * 384 + 256 + head * DH);
        float wgt = lg[a];
#pragma unroll
        for (int d4 = 0; d4 < 8; d4++) {
            float4 vv = vp[d4];
            o[4*d4]   += wgt * vv.x; o[4*d4+1] += wgt * vv.y;
            o[4*d4+2] += wgt * vv.z; o[4*d4+3] += wgt * vv.w;
        }
        aj++; if (aj == KSZ) { aj = 0; ai++; }
    }
    unsigned short* op = attnh + (size_t)pix * CCH + head * DH;
#pragma unroll
    for (int d = 0; d < DH; d++) op[d] = f2bf(o[d] * inv);
}

// ---- global mean of t over H,W per (n,c) -------------------------------
__global__ void k_tmean(const float* __restrict__ t, float* __restrict__ graw) {
    int nc = blockIdx.x; int n = nc >> 7, c = nc & 127;
    int tid = threadIdx.x;
    float s = 0.0f;
    for (int hw = tid; hw < HWW; hw += 256)
        s += t[(((size_t)n << 12) + hw) * CCH + c];
    __shared__ float sm[256];
    sm[tid] = s; __syncthreads();
    for (int o = 128; o > 0; o >>= 1) {
        if (tid < o) sm[tid] += sm[tid + o];
        __syncthreads();
    }
    if (tid == 0) graw[nc] = sm[0] * (1.0f / HWW);
}

// ---- channel-attention squeeze MLP: 128 -> 7 -> 128 --------------------
__global__ void k_ca(const float* __restrict__ graw, const float* __restrict__ ca1_w,
                     const float* __restrict__ ca1_b, const float* __restrict__ ca2_w,
                     const float* __restrict__ ca2_b, float* __restrict__ g) {
    int n = blockIdx.x, c = threadIdx.x;
    __shared__ float sg[CCH]; __shared__ float h[7];
    sg[c] = graw[n * CCH + c]; __syncthreads();
    if (c < 7) {
        float s = ca1_b[c];
        for (int k = 0; k < CCH; k++) s += sg[k] * ca1_w[k * 7 + c];
        h[c] = s > 0.0f ? s : 0.0f;
    }
    __syncthreads();
    float s = ca2_b[c];
#pragma unroll
    for (int r = 0; r < 7; r++) s += h[r] * ca2_w[r * CCH + c];
    g[n * CCH + c] = 1.0f / (1.0f + expf(-s));
}

// ---- x2 = shortcut + na + 0.02*t*g, fused with LN2 ---------------------
__global__ void k_x2ln2(const float* __restrict__ xT, const float* __restrict__ proj,
                        const float* __restrict__ t, const float* __restrict__ g,
                        const float* __restrict__ w, const float* __restrict__ b,
                        float* __restrict__ x2, unsigned short* __restrict__ ln2h) {
    int p = blockIdx.x, c = threadIdx.x, n = p >> 12;
    size_t idx = (size_t)p * CCH + c;
    float v = xT[idx] + proj[idx] + 0.02f * t[idx] * g[n * CCH + c];
    x2[idx] = v;
    __shared__ float s1[CCH], s2[CCH];
    s1[c] = v; s2[c] = v * v; __syncthreads();
    for (int o = 64; o > 0; o >>= 1) {
        if (c < o) { s1[c] += s1[c + o]; s2[c] += s2[c + o]; }
        __syncthreads();
    }
    float mu  = s1[0] * (1.0f / CCH);
    float var = s2[0] * (1.0f / CCH) - mu * mu;
    ln2h[idx] = f2bf((v - mu) * rsqrtf(var + 1e-5f) * w[c] + b[c]);
}

extern "C" void kernel_launch(void* const* d_in, const int* in_sizes, int n_in,
                              void* d_out, int out_size, void* d_ws, size_t ws_size,
                              hipStream_t stream) {
    (void)in_sizes; (void)n_in; (void)out_size; (void)ws_size;
    const float* x      = (const float*)d_in[0];
    const float* ln1_w  = (const float*)d_in[1];
    const float* ln1_b  = (const float*)d_in[2];
    const float* ln2_w  = (const float*)d_in[3];
    const float* ln2_b  = (const float*)d_in[4];
    const float* qkv_w  = (const float*)d_in[5];
    const float* qkv_b  = (const float*)d_in[6];
    const float* proj_w = (const float*)d_in[7];
    const float* proj_b = (const float*)d_in[8];
    const float* rpb    = (const float*)d_in[9];
    const float* conv1_w= (const float*)d_in[10];
    const float* conv1_b= (const float*)d_in[11];
    const float* conv2_w= (const float*)d_in[12];
    const float* conv2_b= (const float*)d_in[13];
    const float* ca1_w  = (const float*)d_in[14];
    const float* ca1_b  = (const float*)d_in[15];
    const float* ca2_w  = (const float*)d_in[16];
    const float* ca2_b  = (const float*)d_in[17];
    const float* fc1_w  = (const float*)d_in[18];
    const float* fc1_b  = (const float*)d_in[19];
    const float* fc2_w  = (const float*)d_in[20];
    const float* fc2_b  = (const float*)d_in[21];

    char* base = (char*)d_ws; size_t off = 0;
    auto alloc = [&](size_t bytes) -> void* {
        void* p = base + off; off = (off + bytes + 255) & ~(size_t)255; return p;
    };
    float* xT             = (float*)alloc((size_t)MTOT * CCH * 4);
    unsigned short* xnh   = (unsigned short*)alloc((size_t)MTOT * CCH * 2);
    float* qkv            = (float*)alloc((size_t)MTOT * 384 * 4);
    unsigned short* t1h   = (unsigned short*)alloc((size_t)MTOT * 32 * 2);
    float* t              = (float*)alloc((size_t)MTOT * CCH * 4);
    float* graw           = (float*)alloc(NB * CCH * 4);
    float* g              = (float*)alloc(NB * CCH * 4);
    unsigned short* attnh = (unsigned short*)alloc((size_t)MTOT * CCH * 2);
    float* projo          = (float*)alloc((size_t)MTOT * CCH * 4);
    float* x2             = (float*)alloc((size_t)MTOT * CCH * 4);
    unsigned short* ln2h  = (unsigned short*)alloc((size_t)MTOT * CCH * 2);
    unsigned short* fc1h  = (unsigned short*)alloc((size_t)MTOT * 512 * 2);
    unsigned short* qkvT  = (unsigned short*)alloc(384 * 128 * 2);
    unsigned short* c1T   = (unsigned short*)alloc(32 * 128 * 2);
    unsigned short* c2T   = (unsigned short*)alloc(128 * 32 * 2);
    unsigned short* projT = (unsigned short*)alloc(128 * 128 * 2);
    unsigned short* fc1T  = (unsigned short*)alloc(512 * 128 * 2);
    unsigned short* fc2T  = (unsigned short*)alloc(128 * 512 * 2);

    auto cw = [&](const float* in, unsigned short* out, int K, int N) {
        int tot = K * N;
        k_convert_wT<<<(tot + 255) / 256, 256, 0, stream>>>(in, out, K, N);
    };
    cw(qkv_w, qkvT, 128, 384);
    cw(conv1_w, c1T, 128, 32);
    cw(conv2_w, c2T, 32, 128);
    cw(proj_w, projT, 128, 128);
    cw(fc1_w, fc1T, 128, 512);
    cw(fc2_w, fc2T, 512, 128);

    k_ln1<<<MTOT, CCH, 0, stream>>>(x, ln1_w, ln1_b, xT, xnh);

    dim3 blk(256);
    // qkv = xn @ qkv_w + b                         (M x 384)
    k_wmma_gemm<<<dim3(384/64, MTOT/128), blk, 0, stream>>>(
        xnh, qkvT, qkv_b, nullptr, qkv, nullptr, 128, 384, 0);
    // t1 = gelu(xn @ conv1_w + b)                  (M x 32, bf16)
    k_wmma_gemm<<<dim3(1, MTOT/128), blk, 0, stream>>>(
        xnh, c1T, conv1_b, nullptr, nullptr, t1h, 128, 32, 1);
    // t = t1 @ conv2_w + b                         (M x 128)
    k_wmma_gemm<<<dim3(2, MTOT/128), blk, 0, stream>>>(
        t1h, c2T, conv2_b, nullptr, t, nullptr, 32, 128, 0);
    // channel attention gate
    k_tmean<<<NB * CCH, 256, 0, stream>>>(t, graw);
    k_ca<<<NB, CCH, 0, stream>>>(graw, ca1_w, ca1_b, ca2_w, ca2_b, g);
    // neighborhood attention -> bf16 pre-proj
    k_na2d<<<MTOT * NHEADS / 256, 256, 0, stream>>>(qkv, rpb, attnh);
    // na_x = attn_out @ proj_w + b                 (M x 128)
    k_wmma_gemm<<<dim3(2, MTOT/128), blk, 0, stream>>>(
        attnh, projT, proj_b, nullptr, projo, nullptr, 128, 128, 0);
    // x2 + LN2
    k_x2ln2<<<MTOT, CCH, 0, stream>>>(xT, projo, t, g, ln2_w, ln2_b, x2, ln2h);
    // m1 = gelu(ln2 @ fc1_w + b)                   (M x 512, bf16)
    k_wmma_gemm<<<dim3(512/64, MTOT/128), blk, 0, stream>>>(
        ln2h, fc1T, fc1_b, nullptr, nullptr, fc1h, 128, 512, 1);
    // out = x2 + m1 @ fc2_w + b, stored NCHW into d_out
    k_wmma_gemm<<<dim3(2, MTOT/128), blk, 0, stream>>>(
        fc1h, fc2T, fc2_b, x2, (float*)d_out, nullptr, 512, 128, 2 | 4);
}
